// NeuralNetwork_29515015258475
// MI455X (gfx1250) — compile-verified, matrix operand order checked by reading, not run on hardware
//
#include <hip/hip_runtime.h>

typedef __attribute__((ext_vector_type(16))) _Float16 v16h;
typedef __attribute__((ext_vector_type(8)))  _Float16 v8h;
typedef __attribute__((ext_vector_type(4)))  _Float16 v4h;
typedef __attribute__((ext_vector_type(8)))  float    v8f;

#define LDSTRIDE 72                 // halves per LDS row (144B: 4-dword bank skew)
#define WAVES 8
#define TILES_PER_WAVE 8
#define ROWS_PER_BLOCK (WAVES * TILES_PER_WAVE * 16)   // 1024 rows per block

// Build a 16-half A/B operand from an LDS row: 8 contiguous halves at p,
// 8 contiguous halves at p+16 (CDNA5 16-bit 16x32 A/B layout:
// lanes<16 -> K {0..7,16..23}, lanes>=16 -> K {8..15,24..31}).
__device__ __forceinline__ v16h make_ab(const _Float16* p) {
  v8h lo = *(const v8h*)p;
  v8h hi = *(const v8h*)(p + 16);
  return __builtin_shufflevector(lo, hi, 0,1,2,3,4,5,6,7,8,9,10,11,12,13,14,15);
}

__global__ __launch_bounds__(256, 2)
void mlp28_fused_kernel(const float* __restrict__ x,
                        const float* __restrict__ W1, const float* __restrict__ b1,
                        const float* __restrict__ W2, const float* __restrict__ b2,
                        const float* __restrict__ W3, const float* __restrict__ b3,
                        float* __restrict__ out, int Btot, int G)
{
  __shared__ _Float16 sW1[64 * LDSTRIDE];     // W1^T: [n][k], f16
  __shared__ _Float16 sW2[32 * LDSTRIDE];     // W2^T: [n][perm(k)], f16
  __shared__ float    sB1[64];
  __shared__ float    sB2[32];
  __shared__ float    sW3v[32];
  __shared__ float    sB3;
  __shared__ _Float16 sX[WAVES][16 * LDSTRIDE];   // per-wave x tile, f16 [m][k]
  __shared__ _Float16 sH[WAVES][16 * LDSTRIDE];   // per-wave h1 tile, f16 [m][perm(k)]
  __shared__ float    sO[WAVES][16];

  const int tid  = threadIdx.x;
  const int wave = tid >> 5;
  const int lane = tid & 31;
  const int g    = blockIdx.y;

  // ---- stage weights (transposed, f16) + biases in LDS, once per block ----
  const float* W1g = W1 + (size_t)g * 64 * 64;      // [i=k][j=n]
  for (int idx = tid; idx < 64 * 64; idx += 256) {
    int i = idx >> 6, j = idx & 63;
    sW1[j * LDSTRIDE + i] = (_Float16)W1g[idx];
  }
  // W2: permute the K axis by pi^-1(i) = (i%16)*4 + i/16 to match the packed
  // h1 restage layout (WMMA contraction is invariant to a shared K-permutation).
  const float* W2g = W2 + (size_t)g * 64 * 32;      // [i=k][j=n]
  for (int idx = tid; idx < 64 * 32; idx += 256) {
    int i = idx >> 5, j = idx & 31;
    int kp = (i & 15) * 4 + (i >> 4);
    sW2[j * LDSTRIDE + kp] = (_Float16)W2g[idx];
  }
  if (tid < 64) sB1[tid] = b1[g * 64 + tid];
  if (tid < 32) { sB2[tid] = b2[g * 32 + tid]; sW3v[tid] = W3[g * 32 + tid]; }
  if (tid == 0) sB3 = b3[g];
  __syncthreads();

  const int klo = (lane < 16) ? 0 : 8;
  const int nn  = lane & 15;
  const int mhi = (lane < 16) ? 0 : 8;

  // ---- pre-build persistent B operands in registers (reused every tile) ----
  v16h w1B[2][4];
  #pragma unroll
  for (int kk = 0; kk < 2; ++kk)
    #pragma unroll
    for (int nt = 0; nt < 4; ++nt)
      w1B[kk][nt] = make_ab(&sW1[(nt * 16 + nn) * LDSTRIDE + kk * 32 + klo]);

  v16h w2B[2][2];
  #pragma unroll
  for (int kk = 0; kk < 2; ++kk)
    #pragma unroll
    for (int nt = 0; nt < 2; ++nt)
      w2B[kk][nt] = make_ab(&sW2[(nt * 16 + nn) * LDSTRIDE + kk * 32 + klo]);

  float bb1[4];
  #pragma unroll
  for (int nt = 0; nt < 4; ++nt) bb1[nt] = sB1[nt * 16 + nn];
  float bb2[2];
  #pragma unroll
  for (int nt = 0; nt < 2; ++nt) bb2[nt] = sB2[nt * 16 + nn];
  const float w3a = sW3v[nn];
  const float w3b = sW3v[16 + nn];
  const float b3v = sB3;

  _Float16* sx = sX[wave];
  _Float16* sh = sH[wave];

  // ---- software pipeline: preload tile 0 into registers ----
  float4 xr[8];
  {
    const int rb0 = blockIdx.x * ROWS_PER_BLOCK + wave * 16;
    if (rb0 + 16 <= Btot) {
      const float4* xp4 = (const float4*)(x + (size_t)rb0 * 64);
      #pragma unroll
      for (int j = 0; j < 8; ++j) xr[j] = xp4[j * 32 + lane];
    }
  }

  for (int it = 0; it < TILES_PER_WAVE; ++it) {
    const int rowbase = blockIdx.x * ROWS_PER_BLOCK + it * (WAVES * 16) + wave * 16;
    if (rowbase + 16 > Btot) break;                 // wave-uniform guard

    // ---- store current tile (held in regs) to LDS as f16 [m][k] ----
    #pragma unroll
    for (int j = 0; j < 8; ++j) {
      int fi = (j * 32 + lane) * 4;
      int r = fi >> 6, c = fi & 63;
      v4h h;
      h[0] = (_Float16)xr[j].x; h[1] = (_Float16)xr[j].y;
      h[2] = (_Float16)xr[j].z; h[3] = (_Float16)xr[j].w;
      *(v4h*)&sx[r * LDSTRIDE + c] = h;
    }

    // ---- issue next tile's global loads (overlap with WMMA compute) ----
    {
      const int rbn = rowbase + WAVES * 16;
      if (rbn + 16 <= Btot) {
        const float4* xp4 = (const float4*)(x + (size_t)rbn * 64);
        #pragma unroll
        for (int j = 0; j < 8; ++j) xr[j] = xp4[j * 32 + lane];
      }
    }

    // ---- layer 1: [16x64] = [16x64] x [64x64], 2 K-chunks x 4 N-tiles ----
    v8f c1[4] = {};
    #pragma unroll
    for (int kk = 0; kk < 2; ++kk) {
      v16h a = make_ab(&sx[nn * LDSTRIDE + kk * 32 + klo]);
      #pragma unroll
      for (int nt = 0; nt < 4; ++nt)
        c1[nt] = __builtin_amdgcn_wmma_f32_16x16x32_f16(
            false, a, false, w1B[kk][nt], (short)0, c1[nt], false, false);
    }

    // bias + ReLU, restage h1 as f16 with K-permuted packing:
    // logical column n = nt*16+nn  ->  LDS offset nn*4+nt  (one b64 per vgpr)
    #pragma unroll
    for (int v = 0; v < 8; ++v) {
      v4h hh;
      #pragma unroll
      for (int nt = 0; nt < 4; ++nt) {
        float t = fmaxf(c1[nt][v] + bb1[nt], 0.0f);
        hh[nt] = (_Float16)t;
      }
      *(v4h*)&sh[(v + mhi) * LDSTRIDE + nn * 4] = hh;
    }

    // ---- layer 2: [16x32] = [16x64] x [64x32] (shared-permuted K) ----
    v8f c2[2] = {};
    #pragma unroll
    for (int kk = 0; kk < 2; ++kk) {
      v16h a = make_ab(&sh[nn * LDSTRIDE + kk * 32 + klo]);
      #pragma unroll
      for (int nt = 0; nt < 2; ++nt)
        c2[nt] = __builtin_amdgcn_wmma_f32_16x16x32_f16(
            false, a, false, w2B[kk][nt], (short)0, c2[nt], false, false);
    }

    // ---- layer 3: ReLU then 32-wide dot with w3, butterfly within halves ----
    float tv[8];
    #pragma unroll
    for (int v = 0; v < 8; ++v) {
      float h0 = fmaxf(c2[0][v] + bb2[0], 0.0f);
      float h1 = fmaxf(c2[1][v] + bb2[1], 0.0f);
      tv[v] = h0 * w3a + h1 * w3b;
    }
    #pragma unroll
    for (int off = 1; off < 16; off <<= 1) {
      #pragma unroll
      for (int v = 0; v < 8; ++v)
        tv[v] += __shfl_xor(tv[v], off, 32);
    }

    // lanes 0 / 16 hold rows 0..7 / 8..15 reductions -> stage 16 floats in LDS
    if (lane == 0) {
      *(float4*)&sO[wave][0] = make_float4(tv[0], tv[1], tv[2], tv[3]);
      *(float4*)&sO[wave][4] = make_float4(tv[4], tv[5], tv[6], tv[7]);
    }
    if (lane == 16) {
      *(float4*)&sO[wave][8]  = make_float4(tv[0], tv[1], tv[2], tv[3]);
      *(float4*)&sO[wave][12] = make_float4(tv[4], tv[5], tv[6], tv[7]);
    }
    if (lane < 16) {
      float o = sO[wave][lane] + b3v;
      out[(size_t)(rowbase + lane) * G + g] = o;
    }
  }
}

extern "C" void kernel_launch(void* const* d_in, const int* in_sizes, int n_in,
                              void* d_out, int out_size, void* d_ws, size_t ws_size,
                              hipStream_t stream) {
  const float* x  = (const float*)d_in[0];
  const float* W1 = (const float*)d_in[1];
  const float* b1 = (const float*)d_in[2];
  const float* W2 = (const float*)d_in[3];
  const float* b2 = (const float*)d_in[4];
  const float* W3 = (const float*)d_in[5];
  const float* b3 = (const float*)d_in[6];
  float* out = (float*)d_out;

  const int Btot = in_sizes[0] / 64;     // x: [B, 64]
  const int G    = in_sizes[2] / 64;     // b1: [G, 64] -> 28

  dim3 grid((Btot + ROWS_PER_BLOCK - 1) / ROWS_PER_BLOCK, G);
  mlp28_fused_kernel<<<grid, 256, 0, stream>>>(x, W1, b1, W2, b2, W3, b3, out, Btot, G);
}
